// MultiHeadAttention_4810363372563
// MI455X (gfx1250) — matrix operand order repaired and checked by reading, not compiled, not run on hardware
//
#include <hip/hip_runtime.h>
#include <hip/hip_bf16.h>

typedef __attribute__((ext_vector_type(16))) _Float16 v16h;
typedef __attribute__((ext_vector_type(8)))  _Float16 v8h;
typedef __attribute__((ext_vector_type(8)))  float    v8f;

#define DMODEL 512
#define SEQ    1024
#define NBATCH 8
#define NHEAD  8
#define DHEAD  64

#if defined(__has_builtin)
#if __has_builtin(__builtin_amdgcn_global_load_async_to_lds_b128) && \
    __has_builtin(__builtin_amdgcn_s_wait_asynccnt)
#define USE_ASYNC_LDS 1
#endif
#endif

#ifdef USE_ASYNC_LDS
#define AS1 __attribute__((address_space(1)))
#define AS3 __attribute__((address_space(3)))
typedef __attribute__((__vector_size__(4 * sizeof(int)))) int i32x4;
__device__ __forceinline__ void copy16_g2l(const _Float16* g, _Float16* l) {
  __builtin_amdgcn_global_load_async_to_lds_b128(
      (AS1 i32x4*)g, (AS3 i32x4*)l, 0, 0);
}
__device__ __forceinline__ void wait_async_copies() {
  __builtin_amdgcn_s_wait_asynccnt(0);
}
#else
__device__ __forceinline__ void copy16_g2l(const _Float16* g, _Float16* l) {
  *(v8h*)l = *(const v8h*)g;
}
__device__ __forceinline__ void wait_async_copies() {}
#endif

// ---------------------------------------------------------------------------
// Kernel 0: convert the four 512x512 f32 weight matrices to f16 (contiguous)
// ---------------------------------------------------------------------------
__global__ __launch_bounds__(256) void wconv_kernel(
    const float* __restrict__ Wq, const float* __restrict__ Wk,
    const float* __restrict__ Wv, const float* __restrict__ Wo,
    _Float16* __restrict__ Wh)
{
  int m = blockIdx.y;
  const float* src = (m == 0) ? Wq : (m == 1) ? Wk : (m == 2) ? Wv : Wo;
  _Float16* dst = Wh + (size_t)m * DMODEL * DMODEL;
  int idx = (blockIdx.x * 256 + threadIdx.x) * 4;
  float4 f = *(const float4*)&src[idx];
  dst[idx + 0] = (_Float16)f.x;
  dst[idx + 1] = (_Float16)f.y;
  dst[idx + 2] = (_Float16)f.z;
  dst[idx + 3] = (_Float16)f.w;
}

// ---------------------------------------------------------------------------
// Kernel 1: fused QKV projection.  out = x @ W^T + b   (per blockIdx.z)
//   Q -> qh [B][H][S][64] f16,  K -> kh [B][H][S][64] f16,
//   V -> vt [B][H][64][S] f16 (transposed for the PV B-fragment)
// Tile: 128 (M) x 64 (N) per 256-thread block, K-step 64 (8 WMMA / 2 barriers)
// ---------------------------------------------------------------------------
__global__ __launch_bounds__(256) void proj_qkv_kernel(
    const float* __restrict__ xq, const float* __restrict__ xk,
    const float* __restrict__ xv, const _Float16* __restrict__ Wh,
    const float* __restrict__ bq, const float* __restrict__ bk,
    const float* __restrict__ bv,
    _Float16* __restrict__ qh, _Float16* __restrict__ kh,
    _Float16* __restrict__ vt)
{
  const int z = blockIdx.z;
  const float*    x    = (z == 0) ? xq : (z == 1) ? xk : xv;
  const _Float16* W    = Wh + (size_t)z * DMODEL * DMODEL;
  const float*    bias = (z == 0) ? bq : (z == 1) ? bk : bv;

  __shared__ _Float16 As[128][72];   // 128 rows x 64 K (pad 8)
  __shared__ _Float16 Bs[64][72];    // 64 n-rows x 64 K (pad 8)

  const int m0  = blockIdx.x * 128;
  const int n0  = blockIdx.y * 64;
  const int tid = threadIdx.x;
  const int lane = tid & 31, wv = tid >> 5;
  const int hi = lane >> 4, ln = lane & 15;

  const v8f vzero = {0.f, 0.f, 0.f, 0.f, 0.f, 0.f, 0.f, 0.f};
  v8f acc[4];
#pragma unroll
  for (int t = 0; t < 4; ++t) acc[t] = vzero;

  for (int kk = 0; kk < DMODEL; kk += 64) {
    // stage A: 128x64 f32 -> f16 in LDS (2048 float4 chunks, 8/thread)
#pragma unroll
    for (int j = 0; j < 8; ++j) {
      const int i = tid + j * 256;
      const int r = i >> 4, c4 = (i & 15) << 2;
      float4 f = *(const float4*)&x[(size_t)(m0 + r) * DMODEL + kk + c4];
      As[r][c4 + 0] = (_Float16)f.x;
      As[r][c4 + 1] = (_Float16)f.y;
      As[r][c4 + 2] = (_Float16)f.z;
      As[r][c4 + 3] = (_Float16)f.w;
    }
    // stage B: 64x64 f16 (512 16-byte chunks, 2/thread) -- async when possible
#pragma unroll
    for (int j = 0; j < 2; ++j) {
      const int i = tid + j * 256;
      const int r = i >> 3, c8 = (i & 7) << 3;
      copy16_g2l(&W[(size_t)(n0 + r) * DMODEL + kk + c8], &Bs[r][c8]);
    }
    wait_async_copies();
    __syncthreads();

    // A fragment: row = wv*16 + ln, K = e + hi*16
#pragma unroll
    for (int s = 0; s < 2; ++s) {
      v8h alo = *(const v8h*)&As[wv * 16 + ln][s * 32 + hi * 16];
      v8h ahi = *(const v8h*)&As[wv * 16 + ln][s * 32 + hi * 16 + 8];
      v16h a;
#pragma unroll
      for (int e = 0; e < 8; ++e) { a[e] = alo[e]; a[e + 8] = ahi[e]; }

#pragma unroll
      for (int t = 0; t < 4; ++t) {
        // B fragment: col = ln, K = e + hi*16
        v8h blo = *(const v8h*)&Bs[t * 16 + ln][s * 32 + hi * 16];
        v8h bhi = *(const v8h*)&Bs[t * 16 + ln][s * 32 + hi * 16 + 8];
        v16h b;
#pragma unroll
        for (int e = 0; e < 8; ++e) { b[e] = blo[e]; b[e + 8] = bhi[e]; }
        acc[t] = __builtin_amdgcn_wmma_f32_16x16x32_f16(
            false, a, false, b, (short)0, acc[t], false, false);
      }
    }
    __syncthreads();
  }

  // epilogue: bias, scatter to head-major f16 layouts
#pragma unroll
  for (int t = 0; t < 4; ++t) {
    const int ncol = n0 + t * 16 + ln;
    const float bsum = bias[ncol];
    const int h = ncol >> 6, dd = ncol & 63;
#pragma unroll
    for (int r = 0; r < 8; ++r) {
      const int mg = m0 + wv * 16 + hi * 8 + r;
      const int bb = mg >> 10, ss = mg & 1023;
      const float val = acc[t][r] + bsum;
      if (z == 0)
        qh[((size_t)(bb * NHEAD + h) * SEQ + ss) * DHEAD + dd] = (_Float16)val;
      else if (z == 1)
        kh[((size_t)(bb * NHEAD + h) * SEQ + ss) * DHEAD + dd] = (_Float16)val;
      else
        vt[((size_t)(bb * NHEAD + h) * DHEAD + dd) * SEQ + ss] = (_Float16)val;
    }
  }
}

// ---------------------------------------------------------------------------
// Kernel 2: flash-style attention per (b,h). 4 waves/block, 16 queries/wave,
// 64 keys per iteration, online softmax in f32, P round-trips LDS to convert
// C-layout -> A-layout for the PV WMMAs.
// ---------------------------------------------------------------------------
__global__ __launch_bounds__(128) void attn_kernel(
    const _Float16* __restrict__ qh, const _Float16* __restrict__ kh,
    const _Float16* __restrict__ vt, const int* __restrict__ qmask,
    const int* __restrict__ kmask, _Float16* __restrict__ yh)
{
  const int bh = blockIdx.y;
  const int b = bh >> 3, h = bh & 7;
  const int wv = threadIdx.x >> 5, lane = threadIdx.x & 31;
  const int hi = lane >> 4, ln = lane & 15;
  const int q0 = blockIdx.x * 64 + wv * 16;

  __shared__ _Float16 P[4][16][72];        // wave-private prob tiles
  _Float16 (*Pw)[72] = P[wv];

  const _Float16* Qb = qh + (size_t)bh * SEQ * DHEAD;
  const _Float16* Kb = kh + (size_t)bh * SEQ * DHEAD;
  const _Float16* Vb = vt + (size_t)bh * DHEAD * SEQ;

  // Q fragments (A-layout), two 32-wide d-steps, held in registers
  v16h aq[2];
  {
    const _Float16* qr = Qb + (size_t)(q0 + ln) * DHEAD;
#pragma unroll
    for (int s2 = 0; s2 < 2; ++s2) {
      v8h lo = *(const v8h*)&qr[s2 * 32 + hi * 16];
      v8h hh = *(const v8h*)&qr[s2 * 32 + hi * 16 + 8];
#pragma unroll
      for (int e = 0; e < 8; ++e) { aq[s2][e] = lo[e]; aq[s2][e + 8] = hh[e]; }
    }
  }

  const v8f vzero = {0.f, 0.f, 0.f, 0.f, 0.f, 0.f, 0.f, 0.f};
  float m_r[8], l_r[8];
  v8f o[4];
#pragma unroll
  for (int t = 0; t < 4; ++t) o[t] = vzero;
#pragma unroll
  for (int r = 0; r < 8; ++r) { m_r[r] = -3.0e38f; l_r[r] = 0.f; }

  const float scale = 0.125f;  // 1/sqrt(64)

  for (int kb = 0; kb < SEQ; kb += 64) {
    // ---- scores: 4 tiles of 16 keys, K-dim = d (2 x 32) ----
    v8f c[4];
#pragma unroll
    for (int t = 0; t < 4; ++t) {
      c[t] = vzero;
      const int key = kb + t * 16 + ln;
      const _Float16* kr = Kb + (size_t)key * DHEAD;
#pragma unroll
      for (int s2 = 0; s2 < 2; ++s2) {
        v16h bk = *(const v16h*)&kr[s2 * 32 + hi * 16];  // B: K=e+hi*16 contiguous
        c[t] = __builtin_amdgcn_wmma_f32_16x16x32_f16(
            false, aq[s2], false, bk, (short)0, c[t], false, false);
      }
      const float msk = (kmask[b * SEQ + key] != 0) ? 1.0f : 0.0f;
#pragma unroll
      for (int r = 0; r < 8; ++r)
        c[t][r] = (msk != 0.f) ? c[t][r] * scale : -1.0e9f;
    }

    // ---- online softmax (row = element; reduce across 16-lane half) ----
#pragma unroll
    for (int r = 0; r < 8; ++r) {
      float mt = fmaxf(fmaxf(c[0][r], c[1][r]), fmaxf(c[2][r], c[3][r]));
#pragma unroll
      for (int off = 8; off >= 1; off >>= 1)
        mt = fmaxf(mt, __shfl_xor(mt, off, 32));
      const float mnew = fmaxf(m_r[r], mt);
      const float alpha = __expf(m_r[r] - mnew);
      m_r[r] = mnew;
      float psum = 0.f;
#pragma unroll
      for (int t = 0; t < 4; ++t) {
        const float p = __expf(c[t][r] - mnew);
        c[t][r] = p;
        psum += p;
      }
#pragma unroll
      for (int off = 8; off >= 1; off >>= 1)
        psum += __shfl_xor(psum, off, 32);
      l_r[r] = l_r[r] * alpha + psum;
#pragma unroll
      for (int t = 0; t < 4; ++t) o[t][r] *= alpha;
    }

    // ---- P (C-layout) -> LDS -> A-layout f16 ----
#pragma unroll
    for (int t = 0; t < 4; ++t)
#pragma unroll
      for (int r = 0; r < 8; ++r)
        Pw[hi * 8 + r][t * 16 + ln] = (_Float16)c[t][r];
    asm volatile("s_wait_dscnt 0" ::: "memory");  // DS in-order per wave

    // ---- PV: A = P (16q x 64key), B = V^T rows (d contiguous in lanes) ----
#pragma unroll
    for (int s2 = 0; s2 < 2; ++s2) {
      v8h lo = *(const v8h*)&Pw[ln][s2 * 32 + hi * 16];
      v8h hh = *(const v8h*)&Pw[ln][s2 * 32 + hi * 16 + 8];
      v16h ap;
#pragma unroll
      for (int e = 0; e < 8; ++e) { ap[e] = lo[e]; ap[e + 8] = hh[e]; }
#pragma unroll
      for (int t = 0; t < 4; ++t) {
        const _Float16* vr = Vb + (size_t)(t * 16 + ln) * SEQ + kb;
        v16h bv = *(const v16h*)&vr[s2 * 32 + hi * 16];
        o[t] = __builtin_amdgcn_wmma_f32_16x16x32_f16(
            false, ap, false, bv, (short)0, o[t], false, false);
      }
    }
  }

  // ---- epilogue: 1/l, query mask, write concat-head f16 activation ----
#pragma unroll
  for (int r = 0; r < 8; ++r) {
    const int qrow = q0 + hi * 8 + r;
    const float qm = (qmask[b * SEQ + qrow] != 0) ? 1.0f : 0.0f;
    const float inv = qm / l_r[r];
#pragma unroll
    for (int t = 0; t < 4; ++t) {
      const int dd = t * 16 + ln;
      yh[((size_t)(b * SEQ + qrow)) * DMODEL + h * DHEAD + dd] =
          (_Float16)(o[t][r] * inv);
    }
  }
}

// ---------------------------------------------------------------------------
// Kernel 3: output projection + bias + residual -> f32 into d_out
// Tile 128x64, K-step 64, all-f16 staging (async LDS copies when available)
// ---------------------------------------------------------------------------
__global__ __launch_bounds__(256) void proj_o_kernel(
    const _Float16* __restrict__ yh, const _Float16* __restrict__ Wo16,
    const float* __restrict__ bo, const float* __restrict__ resid,
    float* __restrict__ out)
{
  __shared__ _Float16 As[128][72];
  __shared__ _Float16 Bs[64][72];

  const int m0  = blockIdx.x * 128;
  const int n0  = blockIdx.y * 64;
  const int tid = threadIdx.x;
  const int lane = tid & 31, wv = tid >> 5;
  const int hi = lane >> 4, ln = lane & 15;

  const v8f vzero = {0.f, 0.f, 0.f, 0.f, 0.f, 0.f, 0.f, 0.f};
  v8f acc[4];
#pragma unroll
  for (int t = 0; t < 4; ++t) acc[t] = vzero;

  for (int kk = 0; kk < DMODEL; kk += 64) {
    // A: 128x64 f16 = 1024 chunks, 4/thread
#pragma unroll
    for (int j = 0; j < 4; ++j) {
      const int i = tid + j * 256;
      const int r = i >> 3, c8 = (i & 7) << 3;
      copy16_g2l(&yh[(size_t)(m0 + r) * DMODEL + kk + c8], &As[r][c8]);
    }
    // B: 64x64 f16 = 512 chunks, 2/thread
#pragma unroll
    for (int j = 0; j < 2; ++j) {
      const int i = tid + j * 256;
      const int r = i >> 3, c8 = (i & 7) << 3;
      copy16_g2l(&Wo16[(size_t)(n0 + r) * DMODEL + kk + c8], &Bs[r][c8]);
    }
    wait_async_copies();
    __syncthreads();

#pragma unroll
    for (int s = 0; s < 2; ++s) {
      v8h alo = *(const v8h*)&As[wv * 16 + ln][s * 32 + hi * 16];
      v8h ahi = *(const v8h*)&As[wv * 16 + ln][s * 32 + hi * 16 + 8];
      v16h a;
#pragma unroll
      for (int e = 0; e < 8; ++e) { a[e] = alo[e]; a[e + 8] = ahi[e]; }

#pragma unroll
      for (int t = 0; t < 4; ++t) {
        v8h blo = *(const v8h*)&Bs[t * 16 + ln][s * 32 + hi * 16];
        v8h bhi = *(const v8h*)&Bs[t * 16 + ln][s * 32 + hi * 16 + 8];
        v16h b;
#pragma unroll
        for (int e = 0; e < 8; ++e) { b[e] = blo[e]; b[e + 8] = bhi[e]; }
        acc[t] = __builtin_amdgcn_wmma_f32_16x16x32_f16(
            false, a, false, b, (short)0, acc[t], false, false);
      }
    }
    __syncthreads();
  }

#pragma unroll
  for (int t = 0; t < 4; ++t) {
    const int ncol = n0 + t * 16 + ln;
    const float bsum = bo[ncol];
#pragma unroll
    for (int r = 0; r < 8; ++r) {
      const int mg = m0 + wv * 16 + hi * 8 + r;
      out[(size_t)mg * DMODEL + ncol] =
          acc[t][r] + bsum + resid[(size_t)mg * DMODEL + ncol];
    }
  }
}

// ---------------------------------------------------------------------------
// Kernel 4: LayerNorm in-place, one row (512) per wave, 16 elems/lane
// ---------------------------------------------------------------------------
__global__ __launch_bounds__(256) void ln_kernel(
    float* __restrict__ y, const float* __restrict__ gamma,
    const float* __restrict__ beta)
{
  const int row = blockIdx.x * 8 + (threadIdx.x >> 5);
  const int lane = threadIdx.x & 31;
  float* yr = y + (size_t)row * DMODEL;

  float vals[16];
  float s = 0.f, s2 = 0.f;
#pragma unroll
  for (int i = 0; i < 16; ++i) {
    const float v = yr[lane + i * 32];
    vals[i] = v;
    s += v;
    s2 += v * v;
  }
#pragma unroll
  for (int off = 16; off >= 1; off >>= 1) {
    s  += __shfl_xor(s,  off, 32);
    s2 += __shfl_xor(s2, off, 32);
  }
  const float mu  = s * (1.f / DMODEL);
  const float var = s2 * (1.f / DMODEL) - mu * mu;
  const float rstd = rsqrtf(var + 1e-5f);
#pragma unroll
  for (int i = 0; i < 16; ++i) {
    const int cidx = lane + i * 32;
    yr[cidx] = (vals[i] - mu) * rstd * gamma[cidx] + beta[cidx];
  }
}

// ---------------------------------------------------------------------------
extern "C" void kernel_launch(void* const* d_in, const int* in_sizes, int n_in,
                              void* d_out, int out_size, void* d_ws,
                              size_t ws_size, hipStream_t stream)
{
  const float* q     = (const float*)d_in[0];
  const float* k     = (const float*)d_in[1];
  const float* v     = (const float*)d_in[2];
  const int*   qmask = (const int*)d_in[3];
  const int*   kmask = (const int*)d_in[4];
  const float* Wq    = (const float*)d_in[5];
  const float* bq    = (const float*)d_in[6];
  const float* Wk    = (const float*)d_in[7];
  const float* bk    = (const float*)d_in[8];
  const float* Wv    = (const float*)d_in[9];
  const float* bv    = (const float*)d_in[10];
  const float* Wo    = (const float*)d_in[11];
  const float* bo    = (const float*)d_in[12];
  const float* gamma = (const float*)d_in[13];
  const float* beta  = (const float*)d_in[14];
  float* out = (float*)d_out;

  char* ws = (char*)d_ws;
  _Float16* Wh = (_Float16*)ws;                          //  2 MB (4 weights f16)
  _Float16* qh = (_Float16*)(ws + ((size_t)2  << 20));   //  8 MB
  _Float16* kh = (_Float16*)(ws + ((size_t)10 << 20));   //  8 MB
  _Float16* vt = (_Float16*)(ws + ((size_t)18 << 20));   //  8 MB (transposed V)
  _Float16* yh = (_Float16*)(ws + ((size_t)26 << 20));   //  8 MB -> 34 MB total

  wconv_kernel<<<dim3(256, 4), 256, 0, stream>>>(Wq, Wk, Wv, Wo, Wh);
  proj_qkv_kernel<<<dim3(64, 8, 3), 256, 0, stream>>>(q, k, v, Wh, bq, bk, bv,
                                                      qh, kh, vt);
  attn_kernel<<<dim3(16, 64), 128, 0, stream>>>(qh, kh, vt, qmask, kmask, yh);
  proj_o_kernel<<<dim3(64, 8), 256, 0, stream>>>(
      yh, Wh + (size_t)3 * DMODEL * DMODEL, bo, q, out);
  ln_kernel<<<dim3(1024), 256, 0, stream>>>(out, gamma, beta);
}
